// StdNormalMmdImqKMatcher_56727928045837
// MI455X (gfx1250) — compile-verified, matrix operand order checked by reading, not run on hardware
//
#include <hip/hip_runtime.h>
#include <math.h>

// ---- CDNA5 WMMA vector types (probe-confirmed signatures) ----
typedef __attribute__((ext_vector_type(16))) __bf16 v16bf;
typedef __attribute__((ext_vector_type(8)))  __bf16 v8bf;
typedef __attribute__((ext_vector_type(8)))  float  v8f;

#define DIMS   64
#define NROWS  8192
#define IMQ_C  128.0f            // C = 2*D
#define ST_I   256               // NROWS / 32  (32-row super-tiles)
#define ST_JG  32                // NROWS / (32*8) (8 j-super-tiles per block)
#define NPART  (ST_I * ST_JG)    // 8192 partials per matrix

// ---------------------------------------------------------------------------
// Pass 1: split fp32 into bf16 hi/lo pair (2-term compensated split).
// ---------------------------------------------------------------------------
__global__ __launch_bounds__(256) void split_bf16_kernel(
    const float* __restrict__ x, const float* __restrict__ z,
    __bf16* __restrict__ xhi, __bf16* __restrict__ xlo,
    __bf16* __restrict__ zhi, __bf16* __restrict__ zlo) {
  int i = blockIdx.x * 256 + threadIdx.x;   // grid sized exactly: NROWS*DIMS/256
  float xv = x[i];
  __bf16 xh = (__bf16)xv;
  xhi[i] = xh;
  xlo[i] = (__bf16)(xv - (float)xh);
  float zv = z[i];
  __bf16 zh = (__bf16)zv;
  zhi[i] = zh;
  zlo[i] = (__bf16)(zv - (float)zh);
}

// ---------------------------------------------------------------------------
// Pass 2: exact fp32 row norms ||x_i||^2, ||z_i||^2.
// ---------------------------------------------------------------------------
__global__ __launch_bounds__(256) void norms_kernel(
    const float* __restrict__ x, const float* __restrict__ z,
    float* __restrict__ nx, float* __restrict__ nz) {
  int row = blockIdx.x * 256 + threadIdx.x;  // grid = NROWS/256
  const float4* xr = (const float4*)(x + row * DIMS);
  const float4* zr = (const float4*)(z + row * DIMS);
  float sx = 0.f, sz = 0.f;
#pragma unroll
  for (int i = 0; i < DIMS / 4; ++i) {
    float4 a = xr[i];
    sx += a.x * a.x + a.y * a.y + a.z * a.z + a.w * a.w;
    float4 b = zr[i];
    sz += b.x * b.x + b.y * b.y + b.z * b.z + b.w * b.w;
  }
  nx[row] = sx;
  nz[row] = sz;
}

// ---------------------------------------------------------------------------
// bf16 16x32 A/B fragment load, ISA 7.12.2 layout:
// lane (m = lane&15, h = lane>>4) holds K = {32*kb + 8h .. +7} in VGPR0-3
// and K = {32*kb + 16 + 8h .. +7} in VGPR4-7 -> two contiguous b128 loads.
// ---------------------------------------------------------------------------
__device__ __forceinline__ v16bf load_frag(const __bf16* __restrict__ row,
                                           int kb, int h) {
  v8bf c0 = *(const v8bf*)(row + 32 * kb + 8 * h);
  v8bf c1 = *(const v8bf*)(row + 32 * kb + 16 + 8 * h);
  return __builtin_shufflevector(c0, c1, 0, 1, 2, 3, 4, 5, 6, 7,
                                 8, 9, 10, 11, 12, 13, 14, 15);
}

#define WMMA_BF16(A, B, C)                                                 \
  __builtin_amdgcn_wmma_f32_16x16x32_bf16(false, (A), false, (B), (short)0, \
                                          (C), false, false)

// IMQ sum over one 16x16 subtile held in a v8f accumulator.
// C layout: lanes 0-15: VGPR r -> M=r, N=lane; lanes 16-31: M=r+8.
__device__ __forceinline__ float subtile_imq(const v8f& acc,
                                             const float* __restrict__ na,
                                             int row_base, int h, float nbv) {
  const float4* nap = (const float4*)(na + row_base + (h << 3));
  float4 n0 = nap[0];
  float4 n1 = nap[1];
  float nav[8] = {n0.x, n0.y, n0.z, n0.w, n1.x, n1.y, n1.z, n1.w};
  float s = 0.f;
#pragma unroll
  for (int v = 0; v < 8; ++v) {
    float sq = fmaxf(nav[v] + nbv - 2.0f * acc[v], 0.0f);
    // fast reciprocal (v_rcp_f32, ~1 ulp): plenty for a scalar MMD estimate
    s += IMQ_C * __builtin_amdgcn_rcpf(IMQ_C + sq);
  }
  return s;
}

// ---------------------------------------------------------------------------
// Pass 3: tiled Gram + IMQ + reduce. One wave per 32x32 super-tile
// (2x2 WMMA tiles, 2x fragment reuse, 4 independent acc chains).
// Block: 256 threads = 8 waves -> i-super-tile blockIdx.x,
// j-super-tiles blockIdx.y*8 .. +7.  SYMM=1: Kxx upper-triangle weighting.
// ---------------------------------------------------------------------------
template <int SYMM>
__global__ __launch_bounds__(256) void imq_tile_kernel(
    const __bf16* __restrict__ Ahi, const __bf16* __restrict__ Alo,
    const __bf16* __restrict__ Bhi, const __bf16* __restrict__ Blo,
    const float* __restrict__ na, const float* __restrict__ nb,
    float* __restrict__ partials) {
  __shared__ float red[8];
  const int wave = threadIdx.x >> 5;
  const int lane = threadIdx.x & 31;
  const int si = blockIdx.x;              // 0..255
  const int sj = blockIdx.y * 8 + wave;   // 0..255

  float wsum = 0.0f;
  // Wave-uniform skip: whole super-tile strictly below diagonal.
  if (!SYMM || sj >= si) {
    const int m = lane & 15;
    const int h = lane >> 4;
    const int i0 = si * 32;   // rows i0..i0+31
    const int j0 = sj * 32;   // cols j0..j0+31

    const __bf16* arh0 = Ahi + (i0 + m) * DIMS;
    const __bf16* arh1 = Ahi + (i0 + 16 + m) * DIMS;
    const __bf16* arl0 = Alo + (i0 + m) * DIMS;
    const __bf16* arl1 = Alo + (i0 + 16 + m) * DIMS;
    const __bf16* brh0 = Bhi + (j0 + m) * DIMS;
    const __bf16* brh1 = Bhi + (j0 + 16 + m) * DIMS;
    const __bf16* brl0 = Blo + (j0 + m) * DIMS;
    const __bf16* brl1 = Blo + (j0 + 16 + m) * DIMS;

    v8f a00 = {}, a01 = {}, a10 = {}, a11 = {};
#pragma unroll
    for (int kb = 0; kb < 2; ++kb) {      // two K=32 blocks cover D=64
      v16bf ah0 = load_frag(arh0, kb, h);
      v16bf ah1 = load_frag(arh1, kb, h);
      v16bf al0 = load_frag(arl0, kb, h);
      v16bf al1 = load_frag(arl1, kb, h);
      v16bf bh0 = load_frag(brh0, kb, h);
      v16bf bh1 = load_frag(brh1, kb, h);
      v16bf bl0 = load_frag(brl0, kb, h);
      v16bf bl1 = load_frag(brl1, kb, h);
      // hi*hi + hi*lo + lo*hi per output tile; 12 WMMAs per K-block
      a00 = WMMA_BF16(ah0, bh0, a00);
      a01 = WMMA_BF16(ah0, bh1, a01);
      a10 = WMMA_BF16(ah1, bh0, a10);
      a11 = WMMA_BF16(ah1, bh1, a11);
      a00 = WMMA_BF16(ah0, bl0, a00);
      a01 = WMMA_BF16(ah0, bl1, a01);
      a10 = WMMA_BF16(ah1, bl0, a10);
      a11 = WMMA_BF16(ah1, bl1, a11);
      a00 = WMMA_BF16(al0, bh0, a00);
      a01 = WMMA_BF16(al0, bh1, a01);
      a10 = WMMA_BF16(al1, bh0, a10);
      a11 = WMMA_BF16(al1, bh1, a11);
    }

    // Per-subtile symmetric weights (ti = 2*si+a, tj = 2*sj+b).
    float w00 = 1.f, w01 = 1.f, w10 = 1.f, w11 = 1.f;
    if (SYMM) {
      const int ti0 = 2 * si, tj0 = 2 * sj;
      w00 = (tj0     > ti0    ) ? 2.f : ((tj0     == ti0    ) ? 1.f : 0.f);
      w01 = (tj0 + 1 > ti0    ) ? 2.f : ((tj0 + 1 == ti0    ) ? 1.f : 0.f);
      w10 = (tj0     > ti0 + 1) ? 2.f : ((tj0     == ti0 + 1) ? 1.f : 0.f);
      w11 = (tj0 + 1 > ti0 + 1) ? 2.f : ((tj0 + 1 == ti0 + 1) ? 1.f : 0.f);
    }

    const float nb0 = nb[j0 + m];
    const float nb1 = nb[j0 + 16 + m];
    wsum  = w00 * subtile_imq(a00, na, i0,      h, nb0);
    wsum += w01 * subtile_imq(a01, na, i0,      h, nb1);
    wsum += w10 * subtile_imq(a10, na, i0 + 16, h, nb0);
    wsum += w11 * subtile_imq(a11, na, i0 + 16, h, nb1);
  }

  // wave32 butterfly reduce
#pragma unroll
  for (int off = 16; off > 0; off >>= 1) wsum += __shfl_xor(wsum, off, 32);
  if (lane == 0) red[wave] = wsum;
  __syncthreads();
  if (threadIdx.x == 0) {
    float s = 0.f;
#pragma unroll
    for (int w = 0; w < 8; ++w) s += red[w];
    partials[blockIdx.y * gridDim.x + blockIdx.x] = s;
  }
}

// ---------------------------------------------------------------------------
// Pass 4: deterministic fixed-order finalize (double accumulation).
// ---------------------------------------------------------------------------
__global__ __launch_bounds__(256) void finalize_kernel(
    const float* __restrict__ pxx, const float* __restrict__ pxz,
    float* __restrict__ out) {
  __shared__ double s1[256], s2[256];
  double a = 0.0, b = 0.0;
  for (int i = threadIdx.x; i < NPART; i += 256) {
    a += (double)pxx[i];
    b += (double)pxz[i];
  }
  s1[threadIdx.x] = a;
  s2[threadIdx.x] = b;
  __syncthreads();
  for (int off = 128; off > 0; off >>= 1) {
    if (threadIdx.x < off) {
      s1[threadIdx.x] += s1[threadIdx.x + off];
      s2[threadIdx.x] += s2[threadIdx.x + off];
    }
    __syncthreads();
  }
  if (threadIdx.x == 0) {
    double Bn = (double)NROWS;
    double sum_xx_offdiag = s1[0] - Bn;   // diagonal: k(x,x)=1, B entries
    double c_s = 1.0 / (Bn * (Bn - 1.0));
    double c_d = 2.0 / (Bn * Bn);
    double val = c_s * sum_xx_offdiag - c_d * s2[0];
    if (val < 1e-8) val = 1e-8;
    out[0] = (float)sqrt(val);
  }
}

// ---------------------------------------------------------------------------
extern "C" void kernel_launch(void* const* d_in, const int* in_sizes, int n_in,
                              void* d_out, int out_size, void* d_ws, size_t ws_size,
                              hipStream_t stream) {
  const float* x = (const float*)d_in[0];
  const float* z = (const float*)d_in[1];
  float* out = (float*)d_out;

  // Workspace layout (bytes):
  //   nx: 32KB | nz: 32KB | xhi,xlo,zhi,zlo: 1MB each | pxx,pxz: 32KB each
  char* wsb = (char*)d_ws;
  float*  nx  = (float*)(wsb);
  float*  nz  = (float*)(wsb + (32u << 10));
  __bf16* xhi = (__bf16*)(wsb + (64u << 10));
  __bf16* xlo = (__bf16*)(wsb + (64u << 10) + (1u << 20));
  __bf16* zhi = (__bf16*)(wsb + (64u << 10) + (2u << 20));
  __bf16* zlo = (__bf16*)(wsb + (64u << 10) + (3u << 20));
  float*  pxx = (float*)(wsb + (64u << 10) + (4u << 20));
  float*  pxz = pxx + NPART;

  split_bf16_kernel<<<(NROWS * DIMS) / 256, 256, 0, stream>>>(x, z, xhi, xlo, zhi, zlo);
  norms_kernel<<<NROWS / 256, 256, 0, stream>>>(x, z, nx, nz);

  dim3 grid(ST_I, ST_JG);
  // Kxx (symmetric, upper triangle weighted)
  imq_tile_kernel<1><<<grid, 256, 0, stream>>>(xhi, xlo, xhi, xlo, nx, nx, pxx);
  // Kxz (full)
  imq_tile_kernel<0><<<grid, 256, 0, stream>>>(xhi, xlo, zhi, zlo, nx, nz, pxz);

  finalize_kernel<<<1, 256, 0, stream>>>(pxx, pxz, out);
}
